// TimeRnn_3487513444747
// MI455X (gfx1250) — matrix operand order repaired and checked by reading, not compiled, not run on hardware
//
#include <hip/hip_runtime.h>
#include <hip/hip_bf16.h>
#include <math.h>
#include <stdint.h>

// ---------------------------------------------------------------------------
// TimeRNN on MI455X (gfx1250).
//   Both layers reduce to one parallel GEMM (M=65536, N=3x1024, K=1024) plus a
//   cheap elementwise scan c_t = f_t*c_{t-1} + ig_t (gates never depend on c).
//   826 GFLOP of GEMM (~330us at ~2.5 PFLOPS dense bf16 WMMA) vs ~45us of HBM
//   traffic at 23.3 TB/s => WMMA-bound. Therefore the GEMM hot loop must be
//   pure async-copy + ds_load_b128 + v_wmma: all fp32->bf16 conversion is
//   hoisted into one-time prep kernels (x converted once; weights transposed
//   to [n][k] bf16 once, 12MB, L2-resident).
// Workspace (needs ~396MB):
//   [0,128M)    F   : sigmoid(f), bf16
//   [128,256M)  IG  : sigmoid(i)*sigmoid(g), bf16
//   [256,384M)  XC  : bf16 x  (layer-0 A), later overwritten by c0 sequence
//                     (layer-1 A) -- disjoint lifetimes.
//   [384,396M)  WT  : 6 transposed bf16 weight matrices [n][k], 2MB each
// ---------------------------------------------------------------------------

typedef __attribute__((ext_vector_type(16))) __bf16 v16bf;
typedef __attribute__((ext_vector_type(8)))  __bf16 v8bf;
typedef __attribute__((ext_vector_type(4)))  __bf16 v4bf;
typedef __attribute__((ext_vector_type(8)))  float  v8f;
typedef __attribute__((ext_vector_type(4)))  float  v4f;

// Types for the async-to-LDS builtin: (AS1 int4*, AS3 int4*, imm off, cpol)
typedef int v4i_vs __attribute__((vector_size(16)));
typedef __attribute__((address_space(1))) v4i_vs* as1_v4i_ptr;
typedef __attribute__((address_space(3))) v4i_vs* as3_v4i_ptr;

#define SEQ    256
#define BATCH  256
#define DIM    1024
#define MTOT   (SEQ * BATCH)     // 65536
#define BM     64
#define BN     128
#define BK     32
#define KTILES (DIM / BK)        // 32

__device__ __forceinline__ float sigmoidf_fast(float x) {
    return 1.0f / (1.0f + __expf(-x));
}

// 16B global->LDS copy. Preferred: CDNA5 async-to-LDS (ASYNCcnt, no VGPR
// round trip). Fallback: plain load+store (LOADcnt/DScnt, compiler-waited).
__device__ __forceinline__ void copy16_g2l(void* lds, const void* gsrc) {
#if __has_builtin(__builtin_amdgcn_global_load_async_to_lds_b128)
    as1_v4i_ptr g = reinterpret_cast<as1_v4i_ptr>(
        reinterpret_cast<uintptr_t>(gsrc));
    as3_v4i_ptr l = reinterpret_cast<as3_v4i_ptr>(
        (uint32_t)reinterpret_cast<uintptr_t>(lds));
    __builtin_amdgcn_global_load_async_to_lds_b128(g, l, 0, 0);
#else
    *(v8bf*)lds = *(const v8bf*)gsrc;
#endif
}

__device__ __forceinline__ void wait_async_le7() {
    asm volatile("s_wait_asynccnt 0x7" ::: "memory");
}
__device__ __forceinline__ void wait_async_zero() {
    asm volatile("s_wait_asynccnt 0x0" ::: "memory");
}

// ---- one-time prep: fp32 -> bf16 bulk convert (for x) ----------------------
__global__ __launch_bounds__(256)
void cvt_bf16_kernel(const float* __restrict__ src, __bf16* __restrict__ dst) {
    size_t i = ((size_t)blockIdx.x * 256 + threadIdx.x) * 8;
    v4f a = *(const v4f*)(src + i);
    v4f b = *(const v4f*)(src + i + 4);
    v8bf r;
    r[0] = (__bf16)a[0]; r[1] = (__bf16)a[1]; r[2] = (__bf16)a[2]; r[3] = (__bf16)a[3];
    r[4] = (__bf16)b[0]; r[5] = (__bf16)b[1]; r[6] = (__bf16)b[2]; r[7] = (__bf16)b[3];
    *(v8bf*)(dst + i) = r;
}

// ---- one-time prep: W [k][n] fp32 -> Wt [n][k] bf16 (rows 0..1023 only) ----
__global__ __launch_bounds__(256)
void wtrans_kernel(const float* __restrict__ W, __bf16* __restrict__ Wt) {
    __shared__ __bf16 T[32][33];                 // +1 pad vs bank conflicts
    const int k0 = blockIdx.x * 32;
    const int n0 = blockIdx.y * 32;
    const int tid = threadIdx.x;
    {
        int kl = tid >> 3, nl = (tid & 7) * 4;   // coalesced read along n
        v4f w = *(const v4f*)&W[(size_t)(k0 + kl) * DIM + (n0 + nl)];
        T[nl + 0][kl] = (__bf16)w[0];
        T[nl + 1][kl] = (__bf16)w[1];
        T[nl + 2][kl] = (__bf16)w[2];
        T[nl + 3][kl] = (__bf16)w[3];
    }
    __syncthreads();
    {
        int nl = tid >> 3, ks = (tid & 7) * 4;   // coalesced write along k
        v4bf o;
        o[0] = T[nl][ks + 0]; o[1] = T[nl][ks + 1];
        o[2] = T[nl][ks + 2]; o[3] = T[nl][ks + 3];
        *(v4bf*)&Wt[((size_t)(n0 + nl) << 10) + k0 + ks] = o;
    }
}

// ---- main fused 3-gate WMMA GEMM + sigmoid epilogue ------------------------
//   A  : [MTOT][DIM] bf16 row-major, m = t*BATCH + b
//   Wt : [3][DIM n][DIM k] bf16 (transposed weights, gate-major)
//   Wi/Wf/Wg fp32 originals (only row DIM = time column used), bi/bf/bg biases
__global__ __launch_bounds__(256)
void gemm_gates_kernel(const __bf16* __restrict__ A, const __bf16* __restrict__ Wt,
                       const float* __restrict__ Wi, const float* __restrict__ Wf,
                       const float* __restrict__ Wg,
                       const float* __restrict__ bi, const float* __restrict__ bfv,
                       const float* __restrict__ bg,
                       __bf16* __restrict__ Fo, __bf16* __restrict__ IGo) {
    __shared__ __align__(16) __bf16 As[2][BM][BK];        // [buf][m][k]
    __shared__ __align__(16) __bf16 Bs[2][3][BN][BK];     // [buf][gate][n][k]

    const int tid  = threadIdx.x;
    const int lane = tid & 31;
    const int wid  = tid >> 5;       // 0..7
    const int wm   = wid >> 2;       // 0..1
    const int wn   = wid & 3;        // 0..3
    const int half = lane >> 4;
    const int ln   = lane & 15;

    const int n0 = blockIdx.x * BN;
    const int m0 = blockIdx.y * BM;

    const v8f vzero = {};
    v8f acc[3][2][2];
#pragma unroll
    for (int g = 0; g < 3; ++g)
#pragma unroll
        for (int mi = 0; mi < 2; ++mi)
#pragma unroll
            for (int ni = 0; ni < 2; ++ni) acc[g][mi][ni] = vzero;

    // staging addressing: A -> 1 x b128/thread, B -> 6 x b128/thread (7 total)
    const int arow = tid >> 2;          // 0..63
    const int akb  = (tid & 3) * 8;     // 0,8,16,24
    const __bf16* Abase = A + (size_t)(m0 + arow) * DIM + akb;

    auto stage = [&](int k0, int buf) {
        copy16_g2l(&As[buf][arow][akb], Abase + k0);
#pragma unroll
        for (int i = 0; i < 6; ++i) {
            int u  = tid + i * 256;     // 0..1535 units of 16B = 24KB (3 gates)
            int g  = u >> 9;            // gate
            int r  = u & 511;
            int nl = r >> 2;            // 0..127
            int ks = (r & 3) * 8;       // 0,8,16,24
            copy16_g2l(&Bs[buf][g][nl][ks],
                       Wt + ((size_t)g << 20) + ((size_t)(n0 + nl) << 10) + k0 + ks);
        }
    };

    stage(0, 0);
    for (int it = 0; it < KTILES; ++it) {
        const int buf = it & 1;
        if (it + 1 < KTILES) {
            stage((it + 1) * BK, buf ^ 1);   // prefetch next tile into other buf
            wait_async_le7();                // drain current tile (in-order)
        } else {
            wait_async_zero();
        }
        __syncthreads();

        // A fragments: ISA 16-bit A layout -> two contiguous b128 LDS loads.
        v16bf afrag[2];
#pragma unroll
        for (int mi = 0; mi < 2; ++mi) {
            int mrow = wm * 32 + mi * 16 + ln;
            int kb   = half * 8;
            v8bf lo = *(const v8bf*)&As[buf][mrow][kb];
            v8bf hi = *(const v8bf*)&As[buf][mrow][kb + 16];
            afrag[mi] = __builtin_shufflevector(lo, hi, 0, 1, 2, 3, 4, 5, 6, 7,
                                                8, 9, 10, 11, 12, 13, 14, 15);
        }
        // B fragments + 12 v_wmma_f32_16x16x32_bf16 per K step.
#pragma unroll
        for (int g = 0; g < 3; ++g) {
#pragma unroll
            for (int ni = 0; ni < 2; ++ni) {
                int nrow = wn * 32 + ni * 16 + ln;
                int kb   = half * 8;
                v8bf lo = *(const v8bf*)&Bs[buf][g][nrow][kb];
                v8bf hi = *(const v8bf*)&Bs[buf][g][nrow][kb + 16];
                v16bf bfrag = __builtin_shufflevector(lo, hi, 0, 1, 2, 3, 4, 5, 6, 7,
                                                      8, 9, 10, 11, 12, 13, 14, 15);
#pragma unroll
                for (int mi = 0; mi < 2; ++mi) {
                    acc[g][mi][ni] = __builtin_amdgcn_wmma_f32_16x16x32_bf16(
                        false, afrag[mi], false, bfrag, (short)0, acc[g][mi][ni],
                        false, false);
                }
            }
        }
        __syncthreads();
    }

    // epilogue: + bias + (t/256)*W[row DIM], sigmoids, store f and i*g (bf16).
    const float* W3[3] = {Wi, Wf, Wg};
    const float* B3[3] = {bi, bfv, bg};
#pragma unroll
    for (int ni = 0; ni < 2; ++ni) {
        int n = n0 + wn * 32 + ni * 16 + ln;
        float badd[3], wlast[3];
#pragma unroll
        for (int g = 0; g < 3; ++g) {
            badd[g]  = B3[g][n];
            wlast[g] = W3[g][(size_t)DIM * DIM + n];
        }
#pragma unroll
        for (int mi = 0; mi < 2; ++mi) {
#pragma unroll
            for (int r = 0; r < 8; ++r) {
                int m = m0 + wm * 32 + mi * 16 + r + half * 8;
                float tv = (float)(m >> 8) * (1.0f / 256.0f);   // t / TIME_LENGTH
                float ai = acc[0][mi][ni][r] + badd[0] + tv * wlast[0];
                float af = acc[1][mi][ni][r] + badd[1] + tv * wlast[1];
                float ag = acc[2][mi][ni][r] + badd[2] + tv * wlast[2];
                float fs = sigmoidf_fast(af);
                float ig = sigmoidf_fast(ai) * sigmoidf_fast(ag);
                size_t o = (size_t)m * DIM + n;
                Fo[o]  = (__bf16)fs;
                IGo[o] = (__bf16)ig;
            }
        }
    }
}

// ---- elementwise linear scan: c_t = f_t*c_{t-1} + ig_t (fp32 accum) --------
__global__ __launch_bounds__(256)
void scan_kernel(const __bf16* __restrict__ F, const __bf16* __restrict__ IG,
                 __bf16* __restrict__ cseq, float* __restrict__ cfin) {
    const int ch = (blockIdx.x * 256 + threadIdx.x) * 4;
    float c0 = 0.f, c1 = 0.f, c2 = 0.f, c3 = 0.f;
    for (int t = 0; t < SEQ; ++t) {
        size_t idx = (size_t)t * (BATCH * DIM) + ch;
        v4bf f4 = *(const v4bf*)&F[idx];
        v4bf g4 = *(const v4bf*)&IG[idx];
        c0 = (float)f4[0] * c0 + (float)g4[0];
        c1 = (float)f4[1] * c1 + (float)g4[1];
        c2 = (float)f4[2] * c2 + (float)g4[2];
        c3 = (float)f4[3] * c3 + (float)g4[3];
        if (cseq) {
            v4bf cc;
            cc[0] = (__bf16)c0; cc[1] = (__bf16)c1;
            cc[2] = (__bf16)c2; cc[3] = (__bf16)c3;
            *(v4bf*)&cseq[idx] = cc;
        }
    }
    v4f o;
    o[0] = c0; o[1] = c1; o[2] = c2; o[3] = c3;
    *(v4f*)&cfin[ch] = o;
}

extern "C" void kernel_launch(void* const* d_in, const int* in_sizes, int n_in,
                              void* d_out, int out_size, void* d_ws, size_t ws_size,
                              hipStream_t stream) {
    const float* x   = (const float*)d_in[0];
    const float* W0[3] = {(const float*)d_in[1], (const float*)d_in[2], (const float*)d_in[3]};
    const float* bi0 = (const float*)d_in[4];
    const float* bf0 = (const float*)d_in[5];
    const float* bg0 = (const float*)d_in[6];
    const float* W1[3] = {(const float*)d_in[7], (const float*)d_in[8], (const float*)d_in[9]};
    const float* bi1 = (const float*)d_in[10];
    const float* bf1 = (const float*)d_in[11];
    const float* bg1 = (const float*)d_in[12];
    float* out = (float*)d_out;                          // [2, 256, 1024]

    const size_t SEQB = (size_t)MTOT * DIM * sizeof(__bf16);   // 128 MB
    __bf16* F  = (__bf16*)d_ws;
    __bf16* IG = (__bf16*)((char*)d_ws + SEQB);
    __bf16* XC = (__bf16*)((char*)d_ws + 2 * SEQB);      // x-bf16, then c0 seq
    __bf16* WT = (__bf16*)((char*)d_ws + 3 * SEQB);      // 6 x (1<<20) bf16

    // One-time prep: convert x, transpose+convert all six weight matrices.
    cvt_bf16_kernel<<<(MTOT * DIM) / (256 * 8), 256, 0, stream>>>(x, XC);
    dim3 tgrid(DIM / 32, DIM / 32);
    for (int g = 0; g < 3; ++g) {
        wtrans_kernel<<<tgrid, 256, 0, stream>>>(W0[g], WT + ((size_t)g << 20));
        wtrans_kernel<<<tgrid, 256, 0, stream>>>(W1[g], WT + ((size_t)(3 + g) << 20));
    }

    dim3 ggrid(DIM / BN, MTOT / BM);                     // (8, 1024)
    dim3 sgrid((BATCH * DIM) / (256 * 4));               // 256 blocks

    // Layer 0: big parallel GEMM, then scan -> c0 sequence (overwrites x-bf16).
    gemm_gates_kernel<<<ggrid, 256, 0, stream>>>(
        XC, WT, W0[0], W0[1], W0[2], bi0, bf0, bg0, F, IG);
    scan_kernel<<<sgrid, 256, 0, stream>>>(F, IG, XC, out);

    // Layer 1: GEMM on the c0 sequence, scan -> c1 final only.
    gemm_gates_kernel<<<ggrid, 256, 0, stream>>>(
        XC, WT + ((size_t)3 << 20), W1[0], W1[1], W1[2], bi1, bf1, bg1, F, IG);
    scan_kernel<<<sgrid, 256, 0, stream>>>(F, IG, (__bf16*)nullptr,
                                           out + BATCH * DIM);
}
